// LlamaDecoderLayer_36928128811666
// MI455X (gfx1250) — matrix operand order repaired and checked by reading, not compiled, not run on hardware
//
#include <hip/hip_runtime.h>
#include <hip/hip_bf16.h>

// ---------------- types ----------------
typedef __bf16 bf16_t;
typedef bf16_t bf16x8  __attribute__((ext_vector_type(8)));
typedef bf16_t bf16x16 __attribute__((ext_vector_type(16)));
typedef float  floatx8 __attribute__((ext_vector_type(8)));
typedef int    i32x4v  __attribute__((vector_size(16)));   // matches async-LDS builtin param

#define AS1 __attribute__((address_space(1)))
#define AS3 __attribute__((address_space(3)))

// gfx1250 async global->LDS path (guarded; falls back to plain copies)
#if defined(__AMDGCN__) && __has_builtin(__builtin_amdgcn_global_load_async_to_lds_b128)
#define HAVE_ASYNC_LDS 1
#else
#define HAVE_ASYNC_LDS 0
#endif

// ---------------- model dims ----------------
#define HIDDEN 4096
#define NHEADS 32
#define NKV    8
#define HD     128
#define INTER  11008
#define BATCH  2
#define SEQ    2048
#define TOK    (BATCH*SEQ)            // 4096
#define QKVN   ((NHEADS + 2*NKV)*HD)  // 6144

// ---------------- GEMM tiling ----------------
#define BM 128
#define BN 128
#define BK 64
#define LDT 72   // BK + 8 pad (bf16 units) -> 16B-aligned fragment reads

__device__ inline floatx8 zero8() {
  floatx8 v;
  #pragma unroll
  for (int i = 0; i < 8; ++i) v[i] = 0.0f;
  return v;
}

__device__ inline bf16x16 join16(bf16x8 lo, bf16x8 hi) {
  return __builtin_shufflevector(lo, hi, 0,1,2,3,4,5,6,7,8,9,10,11,12,13,14,15);
}

__device__ inline floatx8 wmma_bf16(bf16x16 a, bf16x16 b, floatx8 c) {
  return __builtin_amdgcn_wmma_f32_16x16x32_bf16(false, a, false, b, (short)0, c, false, false);
}

// A fragment (16x32): lane<16 row m=lane%16, ks {0..7,16..23}; lane>=16 ks {8..15,24..31}
__device__ inline bf16x16 frag_a_lds(const bf16_t* s, int stride, int lane, int m_base, int kk) {
  int row  = m_base + (lane & 15);
  int half = lane >> 4;
  const bf16_t* p = s + row * stride + kk * 32 + half * 8;
  return join16(*(const bf16x8*)p, *(const bf16x8*)(p + 16));
}
// B fragment (32x16), LDS n-major: lane n=lane%16, lane<16 ks {0..7,16..23}; lane>=16 ks {8..15,24..31}
__device__ inline bf16x16 frag_b_lds(const bf16_t* s, int stride, int lane, int n_base, int kk) {
  int col  = n_base + (lane & 15);
  int half = lane >> 4;
  const bf16_t* p = s + col * stride + kk * 32 + half * 8;
  return join16(*(const bf16x8*)p, *(const bf16x8*)(p + 16));
}

// 16-byte global -> LDS copy (async on gfx1250 when available)
__device__ inline void copy16_g2l(bf16_t* sp, const bf16_t* gp) {
#if HAVE_ASYNC_LDS
  __builtin_amdgcn_global_load_async_to_lds_b128((AS1 i32x4v*)gp, (AS3 i32x4v*)sp, 0, 0);
#else
  *(bf16x8*)sp = *(const bf16x8*)gp;
#endif
}
__device__ inline void async_copies_done() {
#if HAVE_ASYNC_LDS
#if __has_builtin(__builtin_amdgcn_s_wait_asynccnt)
  __builtin_amdgcn_s_wait_asynccnt(0);
#else
  asm volatile("s_wait_asynccnt 0" ::: "memory");
#endif
#endif
}

// Stage 128 rows x 64 k of bf16 (row-major, leading dim ld) into LDS
__device__ inline void stage_tile(bf16_t* s, const bf16_t* G, int ld, int kmax,
                                  int row0, int k0, int tid) {
  #pragma unroll
  for (int c = 0; c < 4; ++c) {
    int linear = (c * 256 + tid) * 8;
    int row = linear >> 6;        // / BK
    int kc  = linear & 63;
    const bf16_t* gp = G + (size_t)(row0 + row) * (size_t)ld + k0 + kc;
    if (c == 0 && k0 + BK < kmax) __builtin_prefetch(gp + BK, 0, 1);
    copy16_g2l(s + row * LDT + kc, gp);
  }
}

// ---------------- weight convert+transpose: W f32 [K][N] -> Wt bf16 [N][K] ----------------
__global__ __launch_bounds__(256) void convert_wt_kernel(const float* __restrict__ W,
                                                         bf16_t* __restrict__ Wt,
                                                         int K, int N) {
  __shared__ float sT[64][65];
  int k0 = blockIdx.x * 64, n0 = blockIdx.y * 64;
  int t = threadIdx.x;
  #pragma unroll
  for (int c = 0; c < 4; ++c) {
    int linear = (c * 256 + t) * 4;
    int r   = linear >> 6;
    int col = linear & 63;
    float4 v = *(const float4*)(W + (size_t)(k0 + r) * (size_t)N + n0 + col);
    sT[r][col + 0] = v.x; sT[r][col + 1] = v.y;
    sT[r][col + 2] = v.z; sT[r][col + 3] = v.w;
  }
  __syncthreads();
  #pragma unroll
  for (int c = 0; c < 2; ++c) {
    int linear = (c * 256 + t) * 8;
    int n  = linear >> 6;
    int kx = linear & 63;
    bf16x8 o;
    #pragma unroll
    for (int j = 0; j < 8; ++j) o[j] = (bf16_t)sT[kx + j][n];
    *(bf16x8*)(Wt + (size_t)(n0 + n) * (size_t)K + k0 + kx) = o;
  }
}

// ---------------- RMSNorm (f32 in -> bf16 out) ----------------
__global__ __launch_bounds__(256) void rmsnorm_kernel(const float* __restrict__ x,
                                                      const float* __restrict__ w,
                                                      bf16_t* __restrict__ out) {
  __shared__ float red[256];
  int t = blockIdx.x;
  const float* row = x + (size_t)t * HIDDEN;
  float ss = 0.0f;
  for (int i = threadIdx.x; i < HIDDEN; i += 256) { float v = row[i]; ss += v * v; }
  red[threadIdx.x] = ss;
  __syncthreads();
  for (int s = 128; s > 0; s >>= 1) {
    if (threadIdx.x < s) red[threadIdx.x] += red[threadIdx.x + s];
    __syncthreads();
  }
  float inv = rsqrtf(red[0] / (float)HIDDEN + 1e-6f);
  for (int i = threadIdx.x; i < HIDDEN; i += 256)
    out[(size_t)t * HIDDEN + i] = (bf16_t)(row[i] * inv * w[i]);
}

// ---------------- GEMM core (A bf16 [M,K], Wt bf16 [N,K]) ----------------
#define GEMM_PROLOG                                                          \
  __shared__ bf16_t sA[BM * LDT];                                            \
  __shared__ bf16_t sB[BN * LDT];                                            \
  int tid = threadIdx.x, lane = tid & 31, wid = tid >> 5;                    \
  int wm = wid >> 2, wn = wid & 3;                                           \
  int m0 = blockIdx.y * BM, n0 = blockIdx.x * BN;                            \
  floatx8 acc[4][2];                                                         \
  _Pragma("unroll")                                                          \
  for (int mt = 0; mt < 4; ++mt)                                             \
    _Pragma("unroll")                                                        \
    for (int nt = 0; nt < 2; ++nt) acc[mt][nt] = zero8();

#define GEMM_MAINLOOP(Aptr, Bptr, Kdim)                                      \
  for (int k0 = 0; k0 < (Kdim); k0 += BK) {                                  \
    stage_tile(sA, (Aptr), (Kdim), (Kdim), m0, k0, tid);                     \
    stage_tile(sB, (Bptr), (Kdim), (Kdim), n0, k0, tid);                     \
    async_copies_done();                                                     \
    __syncthreads();                                                         \
    _Pragma("unroll")                                                        \
    for (int kk = 0; kk < 2; ++kk) {                                         \
      bf16x16 af[4], bfr[2];                                                 \
      _Pragma("unroll")                                                      \
      for (int mt = 0; mt < 4; ++mt)                                         \
        af[mt] = frag_a_lds(sA, LDT, lane, wm * 64 + mt * 16, kk);           \
      _Pragma("unroll")                                                      \
      for (int nt = 0; nt < 2; ++nt)                                         \
        bfr[nt] = frag_b_lds(sB, LDT, lane, wn * 32 + nt * 16, kk);          \
      _Pragma("unroll")                                                      \
      for (int mt = 0; mt < 4; ++mt)                                         \
        _Pragma("unroll")                                                    \
        for (int nt = 0; nt < 2; ++nt)                                       \
          acc[mt][nt] = wmma_bf16(af[mt], bfr[nt], acc[mt][nt]);             \
    }                                                                        \
    __syncthreads();                                                         \
  }

__global__ __launch_bounds__(256) void gemm_plain_kernel(const bf16_t* __restrict__ A,
                                                         const bf16_t* __restrict__ Wt,
                                                         float* __restrict__ C,
                                                         int N, int K) {
  GEMM_PROLOG
  GEMM_MAINLOOP(A, Wt, K)
  int halfm = (lane >> 4) << 3;
  #pragma unroll
  for (int mt = 0; mt < 4; ++mt)
    #pragma unroll
    for (int nt = 0; nt < 2; ++nt)
      #pragma unroll
      for (int r = 0; r < 8; ++r) {
        size_t gr = (size_t)(m0 + wm * 64 + mt * 16 + halfm + r);
        int gc = n0 + wn * 32 + nt * 16 + (lane & 15);
        C[gr * (size_t)N + gc] = acc[mt][nt][r];
      }
}

__global__ __launch_bounds__(256) void gemm_resid_kernel(const bf16_t* __restrict__ A,
                                                         const bf16_t* __restrict__ Wt,
                                                         const float* __restrict__ R,
                                                         float* __restrict__ C,
                                                         int N, int K) {
  GEMM_PROLOG
  GEMM_MAINLOOP(A, Wt, K)
  int halfm = (lane >> 4) << 3;
  #pragma unroll
  for (int mt = 0; mt < 4; ++mt)
    #pragma unroll
    for (int nt = 0; nt < 2; ++nt)
      #pragma unroll
      for (int r = 0; r < 8; ++r) {
        size_t gr = (size_t)(m0 + wm * 64 + mt * 16 + halfm + r);
        int gc = n0 + wn * 32 + nt * 16 + (lane & 15);
        size_t idx = gr * (size_t)N + gc;
        C[idx] = R[idx] + acc[mt][nt][r];
      }
}

// fused gate/up + SwiGLU: Wt_gu bf16 [2*INTER][HIDDEN], act bf16 [TOK][INTER]
__global__ __launch_bounds__(256) void gemm_gateup_kernel(const bf16_t* __restrict__ A,
                                                          const bf16_t* __restrict__ Wt,
                                                          bf16_t* __restrict__ act,
                                                          int K) {
  __shared__ bf16_t sA[BM * LDT];
  __shared__ bf16_t sBg[BN * LDT];
  __shared__ bf16_t sBu[BN * LDT];
  int tid = threadIdx.x, lane = tid & 31, wid = tid >> 5;
  int wm = wid >> 2, wn = wid & 3;
  int m0 = blockIdx.y * BM, n0 = blockIdx.x * BN;
  floatx8 accg[4][2], accu[4][2];
  #pragma unroll
  for (int mt = 0; mt < 4; ++mt)
    #pragma unroll
    for (int nt = 0; nt < 2; ++nt) { accg[mt][nt] = zero8(); accu[mt][nt] = zero8(); }

  for (int k0 = 0; k0 < K; k0 += BK) {
    stage_tile(sA,  A,  K, K, m0, k0, tid);
    stage_tile(sBg, Wt, K, K, n0, k0, tid);
    stage_tile(sBu, Wt, K, K, INTER + n0, k0, tid);
    async_copies_done();
    __syncthreads();
    #pragma unroll
    for (int kk = 0; kk < 2; ++kk) {
      bf16x16 af[4], bg[2], bu[2];
      #pragma unroll
      for (int mt = 0; mt < 4; ++mt) af[mt] = frag_a_lds(sA, LDT, lane, wm * 64 + mt * 16, kk);
      #pragma unroll
      for (int nt = 0; nt < 2; ++nt) {
        bg[nt] = frag_b_lds(sBg, LDT, lane, wn * 32 + nt * 16, kk);
        bu[nt] = frag_b_lds(sBu, LDT, lane, wn * 32 + nt * 16, kk);
      }
      #pragma unroll
      for (int mt = 0; mt < 4; ++mt)
        #pragma unroll
        for (int nt = 0; nt < 2; ++nt) {
          accg[mt][nt] = wmma_bf16(af[mt], bg[nt], accg[mt][nt]);
          accu[mt][nt] = wmma_bf16(af[mt], bu[nt], accu[mt][nt]);
        }
    }
    __syncthreads();
  }
  int halfm = (lane >> 4) << 3;
  #pragma unroll
  for (int mt = 0; mt < 4; ++mt)
    #pragma unroll
    for (int nt = 0; nt < 2; ++nt)
      #pragma unroll
      for (int r = 0; r < 8; ++r) {
        size_t gr = (size_t)(m0 + wm * 64 + mt * 16 + halfm + r);
        int gc = n0 + wn * 32 + nt * 16 + (lane & 15);
        float g = accg[mt][nt][r], u = accu[mt][nt][r];
        float silu = g / (1.0f + __expf(-g));
        act[gr * (size_t)INTER + gc] = (bf16_t)(silu * u);
      }
}

// ---------------- RoPE + layout split ----------------
__global__ __launch_bounds__(256) void rope_kernel(const float* __restrict__ qkv,
                                                   const int* __restrict__ pos,
                                                   bf16_t* __restrict__ qb,
                                                   bf16_t* __restrict__ kb,
                                                   bf16_t* __restrict__ vt) {
  int idx = blockIdx.x * 256 + threadIdx.x;
  const int total = TOK * 48 * 64;
  if (idx >= total) return;
  int i = idx & 63;
  int tmp = idx >> 6;
  int h = tmp % 48;            // 0..31 q, 32..39 k, 40..47 v
  int tk = tmp / 48;
  int b = tk / SEQ, s = tk % SEQ;
  if (h < 40) {
    const float* base = qkv + (size_t)tk * QKVN + h * HD;
    float x1 = base[i], x2 = base[i + 64];
    float p = (float)pos[s];
    float invf = __powf(10000.0f, -((float)(2 * i)) / 128.0f);
    float fr = p * invf;
    float c = __cosf(fr), sn = __sinf(fr);
    float o1 = x1 * c - x2 * sn;
    float o2 = x2 * c + x1 * sn;
    if (h < 32) {
      bf16_t* d = qb + ((size_t)(b * NHEADS + h) * SEQ + s) * HD;
      d[i] = (bf16_t)o1; d[i + 64] = (bf16_t)o2;
    } else {
      int kh = h - 32;
      bf16_t* d = kb + ((size_t)(b * NKV + kh) * SEQ + s) * HD;
      d[i] = (bf16_t)o1; d[i + 64] = (bf16_t)o2;
    }
  } else {
    int vh = h - 40;
    const float* base = qkv + (size_t)tk * QKVN + (NHEADS + NKV) * HD + vh * HD;
    vt[((size_t)(b * NKV + vh) * HD + i) * SEQ + s]      = (bf16_t)base[i];
    vt[((size_t)(b * NKV + vh) * HD + i + 64) * SEQ + s] = (bf16_t)base[i + 64];
  }
}

// ---------------- flash attention (causal, GQA) ----------------
#define BQ 128
#define BC 64
#define LDK 136   // HD + 8
#define LDV 72    // BC + 8
#define LDP 72

__global__ __launch_bounds__(256) void flash_attn_kernel(const bf16_t* __restrict__ q,
                                                         const bf16_t* __restrict__ k,
                                                         const bf16_t* __restrict__ vt,
                                                         bf16_t* __restrict__ attn) {
  __shared__ bf16_t sK[BC * LDK];        // [kv][d]
  __shared__ bf16_t sV[HD * LDV];        // [d][kv]
  __shared__ bf16_t sP[8 * 16 * LDP];    // per-wave [16 rows][BC]

  int tid = threadIdx.x, lane = tid & 31, w = tid >> 5;
  int half = lane >> 4;
  int qt = blockIdx.x, h = blockIdx.y, b = blockIdx.z;
  int kvh = h >> 2;
  int q0 = qt * BQ;

  bf16x16 qf[4];
  {
    int qrow = q0 + w * 16 + (lane & 15);
    const bf16_t* qbase = q + ((size_t)(b * NHEADS + h) * SEQ + qrow) * HD;
    #pragma unroll
    for (int kk = 0; kk < 4; ++kk) {
      const bf16_t* p = qbase + kk * 32 + half * 8;
      qf[kk] = join16(*(const bf16x8*)p, *(const bf16x8*)(p + 16));
    }
  }

  floatx8 oacc[8];
  #pragma unroll
  for (int d = 0; d < 8; ++d) oacc[d] = zero8();
  float rmax[8], rsum[8];
  #pragma unroll
  for (int r = 0; r < 8; ++r) { rmax[r] = -3.0e38f; rsum[r] = 0.0f; }

  const float scale = 0.08838834764831845f;
  bf16_t* sPw = sP + w * 16 * LDP;
  int ntiles = q0 / BC + 2;

  for (int t = 0; t < ntiles; ++t) {
    int kv0 = t * BC;
    #pragma unroll
    for (int c = 0; c < 4; ++c) {
      int linear = (c * 256 + tid) * 8;
      int kr = linear >> 7;
      int d  = linear & 127;
      copy16_g2l(sK + kr * LDK + d,
                 k + ((size_t)(b * NKV + kvh) * SEQ + kv0 + kr) * HD + d);
    }
    #pragma unroll
    for (int c = 0; c < 4; ++c) {
      int linear = (c * 256 + tid) * 8;
      int d  = linear >> 6;
      int kc = linear & 63;
      copy16_g2l(sV + d * LDV + kc,
                 vt + ((size_t)(b * NKV + kvh) * HD + d) * SEQ + kv0 + kc);
    }
    async_copies_done();
    __syncthreads();

    floatx8 sc[4];
    #pragma unroll
    for (int nt = 0; nt < 4; ++nt) sc[nt] = zero8();
    #pragma unroll
    for (int kk = 0; kk < 4; ++kk) {
      #pragma unroll
      for (int nt = 0; nt < 4; ++nt) {
        bf16x16 bk = frag_b_lds(sK, LDK, lane, nt * 16, kk);
        sc[nt] = wmma_bf16(qf[kk], bk, sc[nt]);
      }
    }

    #pragma unroll
    for (int r = 0; r < 8; ++r) {
      int gq = q0 + w * 16 + (half << 3) + r;
      float mx = rmax[r];
      #pragma unroll
      for (int nt = 0; nt < 4; ++nt) {
        int gk = kv0 + nt * 16 + (lane & 15);
        float sv = sc[nt][r] * scale;
        sv = (gk <= gq) ? sv : -3.0e38f;
        sc[nt][r] = sv;
        mx = fmaxf(mx, sv);
      }
      #pragma unroll
      for (int off = 1; off < 16; off <<= 1) mx = fmaxf(mx, __shfl_xor(mx, off, 32));
      float corr = __expf(rmax[r] - mx);
      rmax[r] = mx;
      float ls = 0.0f;
      #pragma unroll
      for (int nt = 0; nt < 4; ++nt) {
        float pv = __expf(sc[nt][r] - mx);
        ls += pv;
        sPw[((half << 3) + r) * LDP + nt * 16 + (lane & 15)] = (bf16_t)pv;
      }
      #pragma unroll
      for (int off = 1; off < 16; off <<= 1) ls += __shfl_xor(ls, off, 32);
      rsum[r] = rsum[r] * corr + ls;
      #pragma unroll
      for (int d = 0; d < 8; ++d) oacc[d][r] *= corr;
    }

    #pragma unroll
    for (int kk = 0; kk < 2; ++kk) {
      bf16x16 pf = frag_a_lds(sPw, LDP, lane, 0, kk);
      #pragma unroll
      for (int dn = 0; dn < 8; ++dn) {
        bf16x16 vf = frag_b_lds(sV, LDV, lane, dn * 16, kk);
        oacc[dn] = wmma_bf16(pf, vf, oacc[dn]);
      }
    }
    __syncthreads();
  }

  #pragma unroll
  for (int r = 0; r < 8; ++r) {
    float inv = 1.0f / rsum[r];
    size_t row = (size_t)(b * SEQ + q0 + w * 16 + (half << 3) + r);
    #pragma unroll
    for (int dn = 0; dn < 8; ++dn) {
      int col = h * HD + dn * 16 + (lane & 15);
      attn[row * HIDDEN + col] = (bf16_t)(oacc[dn][r] * inv);
    }
  }
}

// ---------------- host launcher ----------------
extern "C" void kernel_launch(void* const* d_in, const int* in_sizes, int n_in,
                              void* d_out, int out_size, void* d_ws, size_t ws_size,
                              hipStream_t stream) {
  const float* x    = (const float*)d_in[0];
  const int*   pos  = (const int*)d_in[1];
  const float* wqkv = (const float*)d_in[2];
  const float* wo   = (const float*)d_in[3];
  const float* wgu  = (const float*)d_in[4];
  const float* wdn  = (const float*)d_in[5];
  const float* ln1  = (const float*)d_in[6];
  const float* ln2  = (const float*)d_in[7];
  float* out = (float*)d_out;

  char* ws = (char*)d_ws;
  size_t off = 0;
  auto alloc = [&](size_t bytes) {
    void* p = ws + off;
    off += (bytes + 255) & ~(size_t)255;
    return p;
  };
  bf16_t* xn     = (bf16_t*)alloc((size_t)TOK * HIDDEN * 2);
  float*  qkv    = (float*) alloc((size_t)TOK * QKVN * 4);
  bf16_t* qb     = (bf16_t*)alloc((size_t)BATCH * NHEADS * SEQ * HD * 2);
  bf16_t* kb     = (bf16_t*)alloc((size_t)BATCH * NKV * SEQ * HD * 2);
  bf16_t* vt     = (bf16_t*)alloc((size_t)BATCH * NKV * SEQ * HD * 2);
  bf16_t* attn   = (bf16_t*)alloc((size_t)TOK * HIDDEN * 2);
  float*  h1     = (float*) alloc((size_t)TOK * HIDDEN * 4);
  bf16_t* xn2    = (bf16_t*)alloc((size_t)TOK * HIDDEN * 2);
  bf16_t* act    = (bf16_t*)alloc((size_t)TOK * INTER * 2);
  bf16_t* wqkv_t = (bf16_t*)alloc((size_t)QKVN * HIDDEN * 2);
  bf16_t* wo_t   = (bf16_t*)alloc((size_t)HIDDEN * HIDDEN * 2);
  bf16_t* wgu_t  = (bf16_t*)alloc((size_t)2 * INTER * HIDDEN * 2);
  bf16_t* wdn_t  = (bf16_t*)alloc((size_t)HIDDEN * INTER * 2);

  // 0. weight f32 [K,N] -> bf16 [N,K]
  convert_wt_kernel<<<dim3(HIDDEN / 64, QKVN / 64), 256, 0, stream>>>(wqkv, wqkv_t, HIDDEN, QKVN);
  convert_wt_kernel<<<dim3(HIDDEN / 64, HIDDEN / 64), 256, 0, stream>>>(wo, wo_t, HIDDEN, HIDDEN);
  convert_wt_kernel<<<dim3(HIDDEN / 64, (2 * INTER) / 64), 256, 0, stream>>>(wgu, wgu_t, HIDDEN, 2 * INTER);
  convert_wt_kernel<<<dim3(INTER / 64, HIDDEN / 64), 256, 0, stream>>>(wdn, wdn_t, INTER, HIDDEN);

  // 1. xn = rmsnorm(x, ln1)
  rmsnorm_kernel<<<TOK, 256, 0, stream>>>(x, ln1, xn);
  // 2. qkv = xn @ w_qkv
  gemm_plain_kernel<<<dim3(QKVN / BN, TOK / BM), 256, 0, stream>>>(xn, wqkv_t, qkv, QKVN, HIDDEN);
  // 3. RoPE + split
  {
    int total = TOK * 48 * 64;
    rope_kernel<<<(total + 255) / 256, 256, 0, stream>>>(qkv, pos, qb, kb, vt);
  }
  // 4. causal flash attention (GQA)
  flash_attn_kernel<<<dim3(SEQ / BQ, NHEADS, BATCH), 256, 0, stream>>>(qb, kb, vt, attn);
  // 5. h1 = x + attn @ w_o
  gemm_resid_kernel<<<dim3(HIDDEN / BN, TOK / BM), 256, 0, stream>>>(attn, wo_t, x, h1, HIDDEN, HIDDEN);
  // 6. xn2 = rmsnorm(h1, ln2)
  rmsnorm_kernel<<<TOK, 256, 0, stream>>>(h1, ln2, xn2);
  // 7. act = silu(xn2 @ w_gate) * (xn2 @ w_up)
  gemm_gateup_kernel<<<dim3(INTER / BN, TOK / BM), 256, 0, stream>>>(xn2, wgu_t, act, HIDDEN);
  // 8. out = h1 + act @ w_down
  gemm_resid_kernel<<<dim3(HIDDEN / BN, TOK / BM), 256, 0, stream>>>(act, wdn_t, h1, out, HIDDEN, INTER);
}
